// TransLayer1_47210280517750
// MI455X (gfx1250) — matrix operand order repaired and checked by reading, not compiled, not run on hardware
//
#include <hip/hip_runtime.h>
#include <math.h>

#define DEV static __device__ __forceinline__

typedef __bf16 v16bf __attribute__((ext_vector_type(16)));
typedef float  v8f   __attribute__((ext_vector_type(8)));
typedef unsigned short ush8 __attribute__((ext_vector_type(8)));

union Frag { v16bf bf; ush8 u[2]; };

DEV float bf2f(unsigned short h){ return __uint_as_float(((unsigned int)h) << 16); }
DEV unsigned short f2bf(float f){
  unsigned int u = __float_as_uint(f);
  u += 0x7fffu + ((u >> 16) & 1u);
  return (unsigned short)(u >> 16);
}

#define BB 4
#define LL 8192
#define DD 512
#define HH 8
#define DHD 64
#define MLM 256
#define SEG 32
#define HID2 2048
#define HS 92
#define HO 46
#define NMRG (BB*HO*HO)   // 8464

// ---------------------------------------------------------------------------
// Generic batched bf16 GEMM:  C = beta*(A @ B) + alpha*I [+bias][+resid][gelu]
// A:  bf16 [M][K] row-major (lda), batch stride sA
// B:  bf16 transposed layout [N][K] (ldb), batch stride sB
// Cf: f32 [M][N]; Cb: bf16 [M][N]; Cbt: bf16 transposed [N][M]
// One wave computes a 32x64 C tile (2 A-frags x 4 B-frags, 8 WMMA / k-step).
// ---------------------------------------------------------------------------
struct GemmP {
  const unsigned short* A; long long sA; int lda;
  const unsigned short* B; long long sB; int ldb;
  float* Cf; int ldcf; long long sC0f, sC1f;
  unsigned short* Cb; int ldcb; long long sC0b, sC1b;
  unsigned short* Cbt; int ldcbt; long long sC1bt;
  int bmod;
  const float* bias;
  const float* resid; long long sR; int ldr;
  int M, N, K;
  float beta, alpha;
  int act;   // 0 none, 1 exact gelu
};

// A-fragment (16x32 bf16), ISA 7.12.2 layout: lane<16 row m=lane holds K[kb..kb+7],K[kb+16..kb+23]
DEV Frag load_a_frag(const unsigned short* base, int ld, int r0, int kb, int pf){
  int lane = threadIdx.x & 31;
  int r = r0 + (lane & 15);
  int hi = lane >> 4;
  const unsigned short* p = base + (long long)r * ld + kb + hi * 8;
  Frag f;
  f.u[0] = *(const ush8*)(p);
  f.u[1] = *(const ush8*)(p + 16);
  if (pf) __builtin_prefetch((const void*)(p + 32), 0, 1);   // next k-step (global_prefetch_b8)
  return f;
}

// B-fragment (32x16 bf16) from Bt [N][K]: lane<16 col n=lane holds K[kb..kb+7],K[kb+16..kb+23]
DEV Frag load_b_frag(const unsigned short* base, int ld, int n0, int kb){
  int lane = threadIdx.x & 31;
  int n = n0 + (lane & 15);
  int hi = lane >> 4;
  const unsigned short* p = base + (long long)n * ld + kb + hi * 8;
  Frag f;
  f.u[0] = *(const ush8*)(p);
  f.u[1] = *(const ush8*)(p + 16);
  return f;
}

__global__ __launch_bounds__(32) void gemm_bf16_kernel(GemmP p){
  int n0 = blockIdx.x * 64;
  int m0 = blockIdx.y * 32;
  int bz = blockIdx.z;
  const unsigned short* A = p.A + (long long)bz * p.sA;
  const unsigned short* B = p.B + (long long)bz * p.sB;
  bool v1 = (m0 + 16) < p.M;          // M is always a multiple of 16
  int m1 = v1 ? (m0 + 16) : m0;

  v8f zero = {0.f,0.f,0.f,0.f,0.f,0.f,0.f,0.f};
  v8f acc[2][4];
  #pragma unroll
  for (int i = 0; i < 2; ++i)
    #pragma unroll
    for (int t = 0; t < 4; ++t) acc[i][t] = zero;

  for (int kb = 0; kb < p.K; kb += 32) {
    int pf = (kb + 32 < p.K);
    Frag fa0 = load_a_frag(A, p.lda, m0, kb, pf);
    Frag fa1 = load_a_frag(A, p.lda, m1, kb, 0);
    Frag fb0 = load_b_frag(B, p.ldb, n0,      kb);
    Frag fb1 = load_b_frag(B, p.ldb, n0 + 16, kb);
    Frag fb2 = load_b_frag(B, p.ldb, n0 + 32, kb);
    Frag fb3 = load_b_frag(B, p.ldb, n0 + 48, kb);
    acc[0][0] = __builtin_amdgcn_wmma_f32_16x16x32_bf16(false, fa0.bf, false, fb0.bf, (short)0, acc[0][0], false, false);
    acc[0][1] = __builtin_amdgcn_wmma_f32_16x16x32_bf16(false, fa0.bf, false, fb1.bf, (short)0, acc[0][1], false, false);
    acc[0][2] = __builtin_amdgcn_wmma_f32_16x16x32_bf16(false, fa0.bf, false, fb2.bf, (short)0, acc[0][2], false, false);
    acc[0][3] = __builtin_amdgcn_wmma_f32_16x16x32_bf16(false, fa0.bf, false, fb3.bf, (short)0, acc[0][3], false, false);
    acc[1][0] = __builtin_amdgcn_wmma_f32_16x16x32_bf16(false, fa1.bf, false, fb0.bf, (short)0, acc[1][0], false, false);
    acc[1][1] = __builtin_amdgcn_wmma_f32_16x16x32_bf16(false, fa1.bf, false, fb1.bf, (short)0, acc[1][1], false, false);
    acc[1][2] = __builtin_amdgcn_wmma_f32_16x16x32_bf16(false, fa1.bf, false, fb2.bf, (short)0, acc[1][2], false, false);
    acc[1][3] = __builtin_amdgcn_wmma_f32_16x16x32_bf16(false, fa1.bf, false, fb3.bf, (short)0, acc[1][3], false, false);
  }

  long long q = bz / p.bmod, rm = bz % p.bmod;
  long long offf = q * p.sC1f + rm * p.sC0f;
  long long offb = q * p.sC1b + rm * p.sC0b;
  long long offbt = (long long)bz * p.sC1bt;
  int lane = threadIdx.x & 31;
  int col = lane & 15;
  int hi8 = (lane >> 4) * 8;
  #pragma unroll
  for (int mi = 0; mi < 2; ++mi) {
    if (mi == 1 && !v1) break;
    #pragma unroll
    for (int t = 0; t < 4; ++t) {
      int n = n0 + t * 16 + col;
      if (n >= p.N) continue;
      #pragma unroll
      for (int r = 0; r < 8; ++r) {
        int m = m0 + mi * 16 + r + hi8;
        float v = p.beta * acc[mi][t][r];
        if (p.alpha != 0.0f && m == n) v += p.alpha;
        if (p.bias)  v += p.bias[n];
        if (p.resid) v += p.resid[(long long)bz * p.sR + (long long)m * p.ldr + n];
        if (p.act == 1) v = 0.5f * v * (1.0f + erff(v * 0.70710678118654752f));
        if (p.Cf)  p.Cf[offf + (long long)m * p.ldcf + n] = v;
        if (p.Cb)  p.Cb[offb + (long long)m * p.ldcb + n] = f2bf(v);
        if (p.Cbt) p.Cbt[offbt + (long long)n * p.ldcbt + m] = f2bf(v);
      }
    }
  }
}

// ---------------------------------------------------------------------------
// block reductions
// ---------------------------------------------------------------------------
DEV float block_sum(float v, float* red){
  int t = threadIdx.x;
  red[t] = v; __syncthreads();
  for (int o = blockDim.x >> 1; o > 0; o >>= 1) { if (t < o) red[t] += red[t + o]; __syncthreads(); }
  float r = red[0]; __syncthreads(); return r;
}
DEV float block_max(float v, float* red){
  int t = threadIdx.x;
  red[t] = v; __syncthreads();
  for (int o = blockDim.x >> 1; o > 0; o >>= 1) { if (t < o) red[t] = fmaxf(red[t], red[t + o]); __syncthreads(); }
  float r = red[0]; __syncthreads(); return r;
}

// weight convert + transpose: in f32 [K][N] -> out bf16 [N][K]
__global__ void cvt_w_t_kernel(const float* __restrict__ in, unsigned short* __restrict__ out, int K, int N){
  long long idx = (long long)blockIdx.x * 256 + threadIdx.x;
  if (idx >= (long long)K * N) return;
  int k = (int)(idx % K);
  int n = (int)(idx / K);
  out[(long long)n * K + k] = f2bf(in[(long long)k * N + n]);
}

// LayerNorm rows (D <= 2048), write bf16
__global__ __launch_bounds__(256) void ln_rows_kernel(const float* __restrict__ x, const float* __restrict__ g,
                                                      const float* __restrict__ b, unsigned short* __restrict__ out, int D){
  __shared__ float red[256];
  long long row = blockIdx.x;
  const float* xr = x + row * D;
  int iters = (D + 255) >> 8;
  float v[8];
  float s = 0.f, s2 = 0.f;
  for (int u = 0; u < iters; ++u) {
    int c = threadIdx.x + (u << 8);
    float val = (c < D) ? xr[c] : 0.f;
    v[u] = val; s += val; s2 += val * val;
  }
  s = block_sum(s, red);
  s2 = block_sum(s2, red);
  float mu = s / (float)D;
  float var = s2 / (float)D - mu * mu;
  float rinv = rsqrtf(var + 1e-5f);
  for (int u = 0; u < iters; ++u) {
    int c = threadIdx.x + (u << 8);
    if (c < D) out[row * D + c] = f2bf((v[u] - mu) * rinv * g[c] + b[c]);
  }
}

// qkv f32 [B*L][1536] -> q_bf (scaled), k_bf [b][h][n][64]; vT_bf [b][h][64][n]
__global__ void repack_qkv_kernel(const float* __restrict__ qkv, unsigned short* __restrict__ qb,
                                  unsigned short* __restrict__ kb, unsigned short* __restrict__ vT){
  long long idx = (long long)blockIdx.x * 256 + threadIdx.x;   // over B*L*512
  if (idx >= (long long)BB * LL * DD) return;
  int c = (int)(idx & 511);
  long long bn = idx >> 9;
  int n = (int)(bn & (LL - 1));
  int b = (int)(bn >> 13);
  int h = c >> 6, d = c & 63;
  long long src = bn * 1536 + c;
  long long dqk = (((long long)(b * HH + h) * LL) + n) * DHD + d;
  qb[dqk] = f2bf(qkv[src] * 0.125f);           // DH^-0.5 = 1/8
  kb[dqk] = f2bf(qkv[src + 512]);
  vT[(((long long)(b * HH + h) * DHD) + d) * LL + n] = f2bf(qkv[src + 1024]);
}

// landmark means: ql (scaled), kl  [b][h][m][64]
__global__ void landmarks_kernel(const float* __restrict__ qkv, unsigned short* __restrict__ ql,
                                 unsigned short* __restrict__ kl){
  long long idx = (long long)blockIdx.x * 256 + threadIdx.x;   // over 32*256*64
  if (idx >= (long long)BB * HH * MLM * DHD) return;
  int d = (int)(idx & 63);
  int m = (int)((idx >> 6) & 255);
  int bh = (int)(idx >> 14);
  int b = bh >> 3, h = bh & 7;
  long long basei = ((long long)b * LL + (long long)m * SEG) * 1536 + h * 64 + d;
  float sq = 0.f, sk = 0.f;
  for (int i = 0; i < SEG; ++i) { sq += qkv[basei + (long long)i * 1536]; sk += qkv[basei + (long long)i * 1536 + 512]; }
  ql[idx] = f2bf(sq * (0.125f / (float)SEG));
  kl[idx] = f2bf(sk * (1.0f / (float)SEG));
}

// in-place bf16 row softmax
__global__ __launch_bounds__(256) void softmax_bf16_kernel(unsigned short* __restrict__ d, int cols){
  __shared__ float red[256];
  long long base = (long long)blockIdx.x * cols;
  float mx = -3.0e38f;
  for (int c = threadIdx.x; c < cols; c += 256) mx = fmaxf(mx, bf2f(d[base + c]));
  mx = block_max(mx, red);
  float s = 0.f;
  for (int c = threadIdx.x; c < cols; c += 256) s += expf(bf2f(d[base + c]) - mx);
  s = block_sum(s, red);
  float inv = 1.0f / s;
  for (int c = threadIdx.x; c < cols; c += 256) d[base + c] = f2bf(expf(bf2f(d[base + c]) - mx) * inv);
}

// pinv init: per-batch max row-sum / col-sum of |a|
__global__ __launch_bounds__(256) void pinv_maxsum_kernel(const unsigned short* __restrict__ a, float* __restrict__ omax){
  __shared__ float red[256];
  const unsigned short* ab = a + (long long)blockIdx.x * (MLM * MLM);
  int t = threadIdx.x;
  float rs = 0.f, cs = 0.f;
  for (int j = 0; j < MLM; ++j) { rs += fabsf(bf2f(ab[t * MLM + j])); cs += fabsf(bf2f(ab[j * MLM + t])); }
  float mr = block_max(rs, red);
  float mc = block_max(cs, red);
  if (t == 0) { omax[blockIdx.x * 2] = mr; omax[blockIdx.x * 2 + 1] = mc; }
}
__global__ void pinv_reduce_kernel(const float* __restrict__ omax, float* __restrict__ invc){
  if (threadIdx.x == 0) {
    float mr = 0.f, mc = 0.f;
    for (int i = 0; i < BB * HH; ++i) { mr = fmaxf(mr, omax[2 * i]); mc = fmaxf(mc, omax[2 * i + 1]); }
    invc[0] = 1.0f / (mr * mc);
  }
}
// z0 = a^T * invc  -> write both layouts: z (A-operand, [M][K]) and zT (B-operand, [N][K])
__global__ void pinv_zinit_kernel(const unsigned short* __restrict__ a, const float* __restrict__ invc,
                                  unsigned short* __restrict__ z, unsigned short* __restrict__ zT){
  long long idx = (long long)blockIdx.x * 256 + threadIdx.x;   // over 32*256*256
  if (idx >= (long long)BB * HH * MLM * MLM) return;
  int bz = (int)(idx >> 16);
  int i = (int)((idx >> 8) & 255);
  int j = (int)(idx & 255);
  float c = invc[0];
  z[idx]  = f2bf(bf2f(a[(long long)bz * (MLM * MLM) + (long long)j * MLM + i]) * c);  // z = a^T c
  zT[idx] = f2bf(bf2f(a[idx]) * c);                                                   // zT = a c
}

// depthwise residual conv over sequence, += into attn_out [(b*L+n)*512 + h*64+d]
__global__ void conv_res_kernel(const unsigned short* __restrict__ vT, const float* __restrict__ kern,
                                float* __restrict__ attn){
  long long idx = (long long)blockIdx.x * 256 + threadIdx.x;   // over B*L*512
  if (idx >= (long long)BB * LL * DD) return;
  int c = (int)(idx & 511);
  long long bn = idx >> 9;
  int n = (int)(bn & (LL - 1));
  int b = (int)(bn >> 13);
  int h = c >> 6, d = c & 63;
  const unsigned short* vp = vT + (((long long)(b * HH + h) * DHD + d) << 13);
  float s = 0.f;
  #pragma unroll
  for (int t = 0; t < 33; ++t) {
    int j = n + t - 16;
    if ((unsigned)j < (unsigned)LL) s += bf2f(vp[j]) * kern[h * 33 + t];
  }
  attn[idx] += s;
}

__global__ void cvt_f32_bf16_kernel(const float* __restrict__ in, unsigned short* __restrict__ out, long long n){
  long long idx = (long long)blockIdx.x * 256 + threadIdx.x;
  if (idx < n) out[idx] = f2bf(in[idx]);
}

// patch merge: gather 2x2 with pad, LN over 2048, write bf16
__global__ __launch_bounds__(256) void merge_ln_kernel(const float* __restrict__ x2, const float* __restrict__ g,
                                                       const float* __restrict__ b, unsigned short* __restrict__ out){
  __shared__ float red[256];
  int row = blockIdx.x;                 // 0..NMRG-1
  int bidx = row / (HO * HO);
  int ij = row % (HO * HO);
  int i = ij / HO, j = ij % HO;
  float v[8];
  float s = 0.f, s2 = 0.f;
  #pragma unroll
  for (int u = 0; u < 8; ++u) {
    int c = threadIdx.x + (u << 8);     // 0..2047
    int quad = c >> 9;
    int d = c & 511;
    int r = 2 * i + (quad & 1);
    int cc = 2 * j + (quad >> 1);
    int l = r * HS + cc;
    float val = (l < LL) ? x2[((long long)bidx * LL + l) * DD + d] : 0.f;
    v[u] = val; s += val; s2 += val * val;
  }
  s = block_sum(s, red);
  s2 = block_sum(s2, red);
  float mu = s / 2048.0f;
  float var = s2 / 2048.0f - mu * mu;
  float rinv = rsqrtf(var + 1e-5f);
  #pragma unroll
  for (int u = 0; u < 8; ++u) {
    int c = threadIdx.x + (u << 8);
    out[(long long)row * 2048 + c] = f2bf((v[u] - mu) * rinv * g[c] + b[c]);
  }
}

// ---------------------------------------------------------------------------
extern "C" void kernel_launch(void* const* d_in, const int* in_sizes, int n_in,
                              void* d_out, int out_size, void* d_ws, size_t ws_size,
                              hipStream_t stream) {
  (void)in_sizes; (void)n_in; (void)out_size; (void)ws_size;
  const float* x      = (const float*)d_in[0];
  const float* norm_g = (const float*)d_in[1];
  const float* norm_b = (const float*)d_in[2];
  const float* w_qkv  = (const float*)d_in[3];
  const float* w_out  = (const float*)d_in[4];
  const float* b_out  = (const float*)d_in[5];
  const float* res_k  = (const float*)d_in[6];
  const float* n2_g   = (const float*)d_in[7];
  const float* n2_b   = (const float*)d_in[8];
  const float* fc1_w  = (const float*)d_in[9];
  const float* fc1_b  = (const float*)d_in[10];
  const float* fc2_w  = (const float*)d_in[11];
  const float* fc2_b  = (const float*)d_in[12];
  const float* pm_g   = (const float*)d_in[13];
  const float* pm_b   = (const float*)d_in[14];
  const float* pm_w   = (const float*)d_in[15];
  float* out = (float*)d_out;

  char* base = (char*)d_ws;
  size_t off = 0;
  auto alloc = [&](size_t bytes)->size_t { size_t o = off; off = (off + bytes + 255) & ~(size_t)255; return o; };

  const long long NTOK = (long long)BB * LL;            // 32768
  const long long BHh  = BB * HH;                       // 32
  const size_t SQ = (size_t)BHh * MLM * MLM * 2;        // 256x256 batched bf16

  size_t o_wqkvT = alloc((size_t)1536 * 512 * 2);
  size_t o_woutT = alloc((size_t)512 * 512 * 2);
  size_t o_fc1T  = alloc((size_t)2048 * 512 * 2);
  size_t o_fc2T  = alloc((size_t)512 * 2048 * 2);
  size_t o_pmrT  = alloc((size_t)1024 * 2048 * 2);
  size_t o_xn    = alloc((size_t)NTOK * DD * 2);
  size_t o_qkv   = alloc((size_t)NTOK * 1536 * 4);      // f32; reused later as MLP hidden bf16
  size_t o_q     = alloc((size_t)BHh * LL * DHD * 2);
  size_t o_k     = alloc((size_t)BHh * LL * DHD * 2);
  size_t o_vT    = alloc((size_t)BHh * LL * DHD * 2);
  size_t o_ql    = alloc((size_t)BHh * MLM * DHD * 2);
  size_t o_kl    = alloc((size_t)BHh * MLM * DHD * 2);
  size_t o_a2    = alloc(SQ);
  size_t o_za    = alloc(SQ);
  size_t o_zaT   = alloc(SQ);
  size_t o_zb    = alloc(SQ);
  size_t o_zbT   = alloc(SQ);
  size_t o_az    = alloc(SQ);
  size_t o_t1T   = alloc(SQ);
  size_t o_t2T   = alloc(SQ);
  size_t o_t3T   = alloc(SQ);
  size_t o_S3    = alloc((size_t)BHh * MLM * LL * 2);
  size_t o_a3vT  = alloc((size_t)BHh * DHD * MLM * 2);
  size_t o_wmT   = alloc((size_t)BHh * DHD * MLM * 2);
  size_t o_S1    = alloc((size_t)BHh * LL * MLM * 2);
  size_t o_attn  = alloc((size_t)NTOK * DD * 4);
  size_t o_attnb = alloc((size_t)NTOK * DD * 2);
  size_t o_x1    = alloc((size_t)NTOK * DD * 4);
  size_t o_hbf   = alloc((size_t)NTOK * DD * 2);
  size_t o_x2    = alloc((size_t)NTOK * DD * 4);
  size_t o_mrg   = alloc((size_t)NMRG * 2048 * 2);
  size_t o_pmax  = alloc(64 * 4);
  size_t o_invc  = alloc(4);
  size_t o_h1    = o_qkv;                               // alias (qkv f32 dead by MLP time)

  unsigned short* wqkvT = (unsigned short*)(base + o_wqkvT);
  unsigned short* woutT = (unsigned short*)(base + o_woutT);
  unsigned short* fc1T  = (unsigned short*)(base + o_fc1T);
  unsigned short* fc2T  = (unsigned short*)(base + o_fc2T);
  unsigned short* pmrT  = (unsigned short*)(base + o_pmrT);
  unsigned short* xn    = (unsigned short*)(base + o_xn);
  float*          qkv   = (float*)(base + o_qkv);
  unsigned short* qbf   = (unsigned short*)(base + o_q);
  unsigned short* kbf   = (unsigned short*)(base + o_k);
  unsigned short* vT    = (unsigned short*)(base + o_vT);
  unsigned short* ql    = (unsigned short*)(base + o_ql);
  unsigned short* kl    = (unsigned short*)(base + o_kl);
  unsigned short* a2    = (unsigned short*)(base + o_a2);
  unsigned short* za    = (unsigned short*)(base + o_za);
  unsigned short* zaT   = (unsigned short*)(base + o_zaT);
  unsigned short* zb    = (unsigned short*)(base + o_zb);
  unsigned short* zbT   = (unsigned short*)(base + o_zbT);
  unsigned short* az    = (unsigned short*)(base + o_az);
  unsigned short* t1T   = (unsigned short*)(base + o_t1T);
  unsigned short* t2T   = (unsigned short*)(base + o_t2T);
  unsigned short* t3T   = (unsigned short*)(base + o_t3T);
  unsigned short* S3    = (unsigned short*)(base + o_S3);
  unsigned short* a3vT  = (unsigned short*)(base + o_a3vT);
  unsigned short* wmT   = (unsigned short*)(base + o_wmT);
  unsigned short* S1    = (unsigned short*)(base + o_S1);
  float*          attn  = (float*)(base + o_attn);
  unsigned short* attnb = (unsigned short*)(base + o_attnb);
  float*          x1    = (float*)(base + o_x1);
  unsigned short* hbf   = (unsigned short*)(base + o_hbf);
  float*          x2    = (float*)(base + o_x2);
  unsigned short* mrg   = (unsigned short*)(base + o_mrg);
  float*          pmax  = (float*)(base + o_pmax);
  float*          invc  = (float*)(base + o_invc);
  unsigned short* h1    = (unsigned short*)(base + o_h1);

  auto run_gemm = [&](const GemmP& p, int batch) {
    dim3 g((unsigned)((p.N + 63) / 64), (unsigned)((p.M + 31) / 32), (unsigned)batch);
    gemm_bf16_kernel<<<g, dim3(32), 0, stream>>>(p);
  };
  auto G = [&]()->GemmP { GemmP p; __builtin_memset(&p, 0, sizeof(p)); p.bmod = 1; p.beta = 1.0f; return p; };

  // 0) convert weights to bf16 transposed [N][K]
  cvt_w_t_kernel<<<(1536 * 512 + 255) / 256, 256, 0, stream>>>(w_qkv, wqkvT, 512, 1536);
  cvt_w_t_kernel<<<(512 * 512 + 255) / 256, 256, 0, stream>>>(w_out, woutT, 512, 512);
  cvt_w_t_kernel<<<(512 * 2048 + 255) / 256, 256, 0, stream>>>(fc1_w, fc1T, 512, 2048);
  cvt_w_t_kernel<<<(2048 * 512 + 255) / 256, 256, 0, stream>>>(fc2_w, fc2T, 2048, 512);
  cvt_w_t_kernel<<<(2048 * 1024 + 255) / 256, 256, 0, stream>>>(pm_w, pmrT, 2048, 1024);

  // 1) LN1
  ln_rows_kernel<<<(unsigned)NTOK, 256, 0, stream>>>(x, norm_g, norm_b, xn, DD);

  // 2) QKV gemm -> f32
  { GemmP p = G(); p.A = xn; p.lda = DD; p.B = wqkvT; p.ldb = DD;
    p.Cf = qkv; p.ldcf = 1536; p.M = (int)NTOK; p.N = 1536; p.K = DD; run_gemm(p, 1); }

  // 3) repack + landmarks
  repack_qkv_kernel<<<(unsigned)((NTOK * DD + 255) / 256), 256, 0, stream>>>(qkv, qbf, kbf, vT);
  landmarks_kernel<<<(unsigned)((BHh * MLM * DHD + 255) / 256), 256, 0, stream>>>(qkv, ql, kl);

  // 4) a2 = softmax(ql @ kl^T)
  { GemmP p = G(); p.A = ql; p.sA = MLM * DHD; p.lda = DHD; p.B = kl; p.sB = MLM * DHD; p.ldb = DHD;
    p.Cb = a2; p.ldcb = MLM; p.sC1b = MLM * MLM; p.M = MLM; p.N = MLM; p.K = DHD; run_gemm(p, (int)BHh); }
  softmax_bf16_kernel<<<(unsigned)(BHh * MLM), 256, 0, stream>>>(a2, MLM);

  // 5) pinv init + 6 Newton iterations (batched 256^3 bf16 WMMA gemms, no transposed-load path)
  pinv_maxsum_kernel<<<(unsigned)BHh, 256, 0, stream>>>(a2, pmax);
  pinv_reduce_kernel<<<1, 32, 0, stream>>>(pmax, invc);
  pinv_zinit_kernel<<<(unsigned)((BHh * MLM * MLM + 255) / 256), 256, 0, stream>>>(a2, invc, za, zaT);

  auto pgemm = [&](const unsigned short* A, const unsigned short* Bt,
                   unsigned short* Cb, unsigned short* Cbt, float beta, float alpha) {
    GemmP p = G(); p.A = A; p.sA = MLM * MLM; p.lda = MLM;
    p.B = Bt; p.sB = MLM * MLM; p.ldb = MLM;
    if (Cb)  { p.Cb = Cb;   p.ldcb = MLM;  p.sC1b = MLM * MLM; }
    if (Cbt) { p.Cbt = Cbt; p.ldcbt = MLM; p.sC1bt = MLM * MLM; }
    p.M = MLM; p.N = MLM; p.K = MLM; p.beta = beta; p.alpha = alpha;
    run_gemm(p, (int)BHh);
  };
  for (int it = 0; it < 6; ++it) {
    unsigned short* zc  = (it & 1) ? zb  : za;
    unsigned short* zcT = (it & 1) ? zbT : zaT;
    unsigned short* zn  = (it & 1) ? za  : zb;
    unsigned short* znT = (it & 1) ? zaT : zbT;
    pgemm(a2, zcT, az,      nullptr, 1.0f,  0.0f);   // az = a @ z
    pgemm(a2, zcT, nullptr, t1T,    -1.0f,  7.0f);   // t1 = 7I - a@z
    pgemm(az, t1T, nullptr, t2T,    -1.0f, 15.0f);   // t2 = 15I - az@t1
    pgemm(az, t2T, nullptr, t3T,    -1.0f, 13.0f);   // t3 = 13I - az@t2
    pgemm(zc, t3T, zn,      znT,     0.25f, 0.0f);   // z' = 0.25 * z@t3
  }
  // final z in za

  // 6) a3 = softmax(ql @ k^T), a3v = a3 @ v  (store a3v transposed for B-use)
  { GemmP p = G(); p.A = ql; p.sA = MLM * DHD; p.lda = DHD; p.B = kbf; p.sB = (long long)LL * DHD; p.ldb = DHD;
    p.Cb = S3; p.ldcb = LL; p.sC1b = (long long)MLM * LL; p.M = MLM; p.N = LL; p.K = DHD; run_gemm(p, (int)BHh); }
  softmax_bf16_kernel<<<(unsigned)(BHh * MLM), 256, 0, stream>>>(S3, LL);
  { GemmP p = G(); p.A = S3; p.sA = (long long)MLM * LL; p.lda = LL; p.B = vT; p.sB = (long long)DHD * LL; p.ldb = LL;
    p.Cbt = a3vT; p.ldcbt = MLM; p.sC1bt = DHD * MLM; p.M = MLM; p.N = DHD; p.K = LL; run_gemm(p, (int)BHh); }

  // 7) W = pinv(a2) @ a3v (store transposed for B-use)
  { GemmP p = G(); p.A = za; p.sA = MLM * MLM; p.lda = MLM; p.B = a3vT; p.sB = DHD * MLM; p.ldb = MLM;
    p.Cbt = wmT; p.ldcbt = MLM; p.sC1bt = DHD * MLM; p.M = MLM; p.N = DHD; p.K = MLM; run_gemm(p, (int)BHh); }

  // 8) a1 = softmax(q @ kl^T); out = a1 @ W  -> attn layout (b, n, h*64+d)
  { GemmP p = G(); p.A = qbf; p.sA = (long long)LL * DHD; p.lda = DHD; p.B = kl; p.sB = MLM * DHD; p.ldb = DHD;
    p.Cb = S1; p.ldcb = MLM; p.sC1b = (long long)LL * MLM; p.M = LL; p.N = MLM; p.K = DHD; run_gemm(p, (int)BHh); }
  softmax_bf16_kernel<<<(unsigned)(BHh * LL), 256, 0, stream>>>(S1, MLM);
  { GemmP p = G(); p.A = S1; p.sA = (long long)LL * MLM; p.lda = MLM; p.B = wmT; p.sB = DHD * MLM; p.ldb = MLM;
    p.Cf = attn; p.ldcf = DD; p.bmod = HH; p.sC0f = DHD; p.sC1f = (long long)LL * DD;
    p.M = LL; p.N = DHD; p.K = MLM; run_gemm(p, (int)BHh); }

  // 9) += depthwise residual conv; then bf16
  conv_res_kernel<<<(unsigned)((NTOK * DD + 255) / 256), 256, 0, stream>>>(vT, res_k, attn);
  cvt_f32_bf16_kernel<<<(unsigned)((NTOK * DD + 255) / 256), 256, 0, stream>>>(attn, attnb, NTOK * DD);

  // 10) x1 = x + attn @ w_out + b_out
  { GemmP p = G(); p.A = attnb; p.lda = DD; p.B = woutT; p.ldb = DD;
    p.bias = b_out; p.resid = x; p.ldr = DD;
    p.Cf = x1; p.ldcf = DD; p.M = (int)NTOK; p.N = DD; p.K = DD; run_gemm(p, 1); }

  // 11) MLP: h = gelu(ln2(x1) @ fc1 + b1); x2 = x1 + h @ fc2 + b2
  ln_rows_kernel<<<(unsigned)NTOK, 256, 0, stream>>>(x1, n2_g, n2_b, hbf, DD);
  { GemmP p = G(); p.A = hbf; p.lda = DD; p.B = fc1T; p.ldb = DD;
    p.bias = fc1_b; p.act = 1;
    p.Cb = h1; p.ldcb = HID2; p.M = (int)NTOK; p.N = HID2; p.K = DD; run_gemm(p, 1); }
  { GemmP p = G(); p.A = h1; p.lda = HID2; p.B = fc2T; p.ldb = HID2;
    p.bias = fc2_b; p.resid = x1; p.ldr = DD;
    p.Cf = x2; p.ldcf = DD; p.M = (int)NTOK; p.N = DD; p.K = HID2; run_gemm(p, 1); }

  // 12) patch merge (gather + pad + LN) then reduction gemm -> d_out
  merge_ln_kernel<<<(unsigned)NMRG, 256, 0, stream>>>(x2, pm_g, pm_b, mrg);
  { GemmP p = G(); p.A = mrg; p.lda = 2048; p.B = pmrT; p.ldb = 2048;
    p.Cf = out; p.ldcf = 1024; p.M = NMRG; p.N = 1024; p.K = 2048; run_gemm(p, 1); }
}
